// RelationalAttentionCross_37606733644543
// MI455X (gfx1250) — matrix operand order repaired and checked by reading, not compiled, not run on hardware
//
#include <hip/hip_runtime.h>

// ---------------------------------------------------------------------------
// RelationalAttentionCross for MI455X (gfx1250, wave32, WMMA bf16 path)
// B=8 N=1024 IN=512 D=64 H=16
//
// Workspace layout (bytes), total ~117 MB:
//   [0,            67108864)  proj  : bf16 [4][B*H][N][D]   (K,Q,V1,V2)
//   [67108864,     67112960)  stats : f32  [4][B*H][2]      (mu, rsqrt)
//   [67112960,     67637248)  m_ws  : f32  [B*H*N]          row max
//   [67637248,     68161536)  l_ws  : f32  [B*H*N]          row sumexp
//   [68161536,     84938752)  E1    : bf16 [B][N][H*D]
//   [84938752,    101715968)  E2    : bf16 [B][N][H*D]
//   [101715968,   110104576)  m1b   : bf16 [B][N][IN]
//   [110104576,   118493184)  m2b   : bf16 [B][N][IN]
//   [118493184,   119541760)  Wkb   : bf16 [H*D][IN]
//   [119541760,   120590336)  Wqb   : bf16 [H*D][IN]
//   [120590336,   121638912)  Wvb   : bf16 [H*D][IN]
//   [121638912,   122687488)  W1b   : bf16 [IN][H*D]
// ---------------------------------------------------------------------------

typedef __attribute__((ext_vector_type(16))) __bf16 v16bf;
typedef __attribute__((ext_vector_type(8)))  __bf16 v8bf;
typedef __attribute__((ext_vector_type(8)))  float  v8f;
typedef __attribute__((ext_vector_type(4)))  float  f4;
typedef __attribute__((ext_vector_type(4)))  unsigned short us4;

static constexpr int kB  = 8;
static constexpr int kN  = 1024;
static constexpr int kIN = 512;
static constexpr int kD  = 64;
static constexpr int kH  = 16;

// Native hardware conversion (v_cvt_*bf16_f32, RNE).
__device__ __forceinline__ unsigned short f2bfu(float f) {
  __bf16 h = (__bf16)f;
  return __builtin_bit_cast(unsigned short, h);
}
__device__ __forceinline__ float bfu2f(unsigned short s) {
  unsigned u = ((unsigned)s) << 16;
  return __builtin_bit_cast(float, u);
}

__device__ __forceinline__ v8f wmma_bf16(v16bf a, v16bf b, v8f c) {
  return __builtin_amdgcn_wmma_f32_16x16x32_bf16(
      false, a, false, b, (short)0, c, false, false);
}

// A-matrix 16x32 bf16 fragment: lane holds row = lane&15,
// elements j -> k = kb+j, kb = (lane<16 ? 0 : 16).
// `p` must point at rowStart + kchunk*32 + kb, and be 16B aligned.
__device__ __forceinline__ v16bf afrag(const unsigned short* p) {
  v8bf lo = *(const v8bf*)p;
  v8bf hi = *(const v8bf*)(p + 8);
  return __builtin_shufflevector(lo, hi, 0, 1, 2, 3, 4, 5, 6, 7,
                                 8, 9, 10, 11, 12, 13, 14, 15);
}

// B-matrix 32x16 bf16 fragment: lane holds col n = lane&15,
// elements j -> k = ((lane<16)?0:16) + j. `p` points at the 16 contiguous
// k-values for this lane's column (16B aligned).
__device__ __forceinline__ v16bf bfrag(const unsigned short* p) {
  v8bf lo = *(const v8bf*)p;
  v8bf hi = *(const v8bf*)(p + 8);
  return __builtin_shufflevector(lo, hi, 0, 1, 2, 3, 4, 5, 6, 7,
                                 8, 9, 10, 11, 12, 13, 14, 15);
}

// ---------------------------------------------------------------------------
// Kernel 0: fp32 -> bf16 bulk conversion (vectorized 4-wide).
// ---------------------------------------------------------------------------
__global__ __launch_bounds__(256) void k_cvt(const float* __restrict__ src,
                                             unsigned short* __restrict__ dst) {
  const size_t i = ((size_t)blockIdx.x * 256 + threadIdx.x) * 4;
  f4 v = *(const f4*)(src + i);
  us4 o;
#pragma unroll
  for (int j = 0; j < 4; ++j) o[j] = f2bfu(v[j]);
  *(us4*)(dst + i) = o;
}

// ---------------------------------------------------------------------------
// Kernel 1: projections + bias + LayerNorm statistics.
// One workgroup per (slot, b, h): GEMM [N=1024, 64] = X[1024,512] x W_h^T,
// slot: 0 K<-m1*Wk  1 Q<-m2*Wq  2 V1<-m1*Wv  3 V2<-m2*Wv
//
// The 64x512 bf16 W slab (64 KB) is staged once in LDS with an XOR 32-byte-
// chunk swizzle (row stride 1024 B would otherwise put all rows in the same
// banks). A compiler memory barrier in the t-loop stops LICM from hoisting
// 64 B-fragments into registers (which previously caused scratch spills).
// ---------------------------------------------------------------------------
__global__ __launch_bounds__(256) void k_proj_ln(
    const unsigned short* __restrict__ m1b, const unsigned short* __restrict__ m2b,
    const unsigned short* __restrict__ Wqb,
    const unsigned short* __restrict__ Wkb,
    const unsigned short* __restrict__ Wvb,
    const float* __restrict__ bq, const float* __restrict__ bk,
    const float* __restrict__ bv,
    unsigned short* __restrict__ proj, float* __restrict__ stats) {
  const int slot = blockIdx.x >> 7;
  const int bh   = blockIdx.x & 127;
  const int b    = bh >> 4;
  const int h    = bh & 15;

  const unsigned short* X = ((slot == 1) || (slot == 3)) ? m2b : m1b;
  X += (size_t)b * kN * kIN;
  const unsigned short* W = (slot == 0) ? Wkb : (slot == 1) ? Wqb : Wvb;
  const float* bias       = (slot == 0) ? bk : (slot == 1) ? bq : bv;

  const int lane  = threadIdx.x & 31;
  const int wave  = threadIdx.x >> 5;
  const int nl    = lane & 15;
  const int half8 = (lane < 16) ? 0 : 8;
  const int ab    = (lane < 16) ? 0 : 16;
  const int kb    = (lane < 16) ? 0 : 16;
  unsigned short* out = proj + (size_t)(slot * 128 + bh) * (kN * kD);

  // 64 KB: W slab, later reused for the LN reduction.
  __shared__ __align__(16) unsigned short wlds[64 * 512];

  // Cooperative swizzled fill: rows h*64..h*64+63 of W are contiguous 64 KB.
  const unsigned short* wsrc = W + (size_t)h * 64 * kIN;
  for (int ci = threadIdx.x; ci < 4096; ci += 256) {
    const int off   = ci * 8;
    const int row   = off >> 9;
    const int col   = off & 511;
    const int chunk = col >> 4;
    const int scol  = ((chunk ^ (row & 31)) << 4) | (col & 15);
    *(v8bf*)&wlds[row * 512 + scol] = *(const v8bf*)(wsrc + off);
  }
  __syncthreads();

  float lsum = 0.f, lsq = 0.f;

  for (int t = 0; t < 8; ++t) {
    asm volatile("" ::: "memory");  // keep fragment loads inside the loop
    const int row0 = wave * 128 + t * 16;
    v8f acc[4] = {};
    for (int kc = 0; kc < 16; ++kc) {
      const v16bf a = afrag(X + (size_t)(row0 + nl) * kIN + kc * 32 + ab);
      const int koff  = kc * 32 + kb;      // multiple of 16 elements
      const int chunk = koff >> 4;
#pragma unroll
      for (int c = 0; c < 4; ++c) {
        const int row = c * 16 + nl;
        const int scol = (chunk ^ (row & 31)) << 4;
        acc[c] = wmma_bf16(a, bfrag(&wlds[row * 512 + scol]), acc[c]);
      }
    }
#pragma unroll
    for (int c = 0; c < 4; ++c) {
      const int d = c * 16 + nl;
      const float bcol = bias[h * 64 + d];
#pragma unroll
      for (int r = 0; r < 8; ++r) {
        const float v = acc[c][r] + bcol;
        lsum += v;
        lsq  += v * v;
        out[(size_t)(row0 + r + half8) * kD + d] = f2bfu(v);
      }
    }
  }

  // Reuse wlds for the block reduction (W slab no longer needed).
  __syncthreads();
  float* red = (float*)wlds;               // [0,256) sums, [256,512) squares
  red[threadIdx.x]       = lsum;
  red[256 + threadIdx.x] = lsq;
  __syncthreads();
  for (int s = 128; s > 0; s >>= 1) {
    if ((int)threadIdx.x < s) {
      red[threadIdx.x]       += red[threadIdx.x + s];
      red[256 + threadIdx.x] += red[256 + threadIdx.x + s];
    }
    __syncthreads();
  }
  if (threadIdx.x == 0) {
    const float inv = 1.f / (float)(kN * kD);
    const float mu  = red[0] * inv;
    const float var = red[256] * inv - mu * mu;
    stats[(slot * 128 + bh) * 2 + 0] = mu;
    stats[(slot * 128 + bh) * 2 + 1] = rsqrtf(var + 1e-5f);
  }
}

// ---------------------------------------------------------------------------
// Kernel 2: apply LayerNorm affine in place on the bf16 projections.
// Slot 1 (Q) additionally folds in the attention scale 1/sqrt(D) = 0.125
// (power of two -> exact in bf16), so the attention passes need no scaling.
// ---------------------------------------------------------------------------
__global__ __launch_bounds__(256) void k_ln_apply(
    unsigned short* __restrict__ proj, const float* __restrict__ stats,
    const float* __restrict__ qn_w, const float* __restrict__ qn_b,
    const float* __restrict__ kn_w, const float* __restrict__ kn_b,
    const float* __restrict__ vn_w, const float* __restrict__ vn_b) {
  size_t e0 = ((size_t)blockIdx.x * 256 + threadIdx.x) * 4;
#pragma unroll
  for (int i = 0; i < 4; ++i) {
    const size_t e   = e0 + i;
    const int  slot  = (int)(e >> 23);
    const int  bh    = (int)((e >> 16) & 127);
    const int  nd    = (int)(e & 65535);
    const float mu   = stats[(slot * 128 + bh) * 2 + 0];
    const float rs   = stats[(slot * 128 + bh) * 2 + 1];
    const float* w   = (slot == 0) ? kn_w : (slot == 1) ? qn_w : vn_w;
    const float* bb  = (slot == 0) ? kn_b : (slot == 1) ? qn_b : vn_b;
    const float sc   = (slot == 1) ? 0.125f : 1.0f;
    const float x    = bfu2f(proj[e]);
    proj[e] = f2bfu(((x - mu) * rs * w[nd] + bb[nd]) * sc);
  }
}

// ---------------------------------------------------------------------------
// Kernel 3: softmax pass 1 — row max and row sumexp (flash-style online).
// Grid: bh*8 + rb; each wave owns a 16-row Q tile.
// ---------------------------------------------------------------------------
__global__ __launch_bounds__(256) void k_pass1(
    const unsigned short* __restrict__ proj,
    float* __restrict__ m_ws, float* __restrict__ l_ws) {
  const int bh = blockIdx.x >> 3;
  const int rb = blockIdx.x & 7;
  const unsigned short* Qp = proj + (size_t)(128 + bh) * (kN * kD);
  const unsigned short* Kp = proj + (size_t)(bh) * (kN * kD);

  const int lane  = threadIdx.x & 31;
  const int wave  = threadIdx.x >> 5;
  const int nl    = lane & 15;
  const int row0  = rb * 128 + wave * 16;
  const int ab    = (lane < 16) ? 0 : 16;
  const int kb    = (lane < 16) ? 0 : 16;

  v16bf qa[2];
#pragma unroll
  for (int dc = 0; dc < 2; ++dc)
    qa[dc] = afrag(Qp + (size_t)(row0 + nl) * kD + dc * 32 + ab);

  float rm[8], rl[8];
#pragma unroll
  for (int r = 0; r < 8; ++r) { rm[r] = -3.0e38f; rl[r] = 0.f; }

  for (int jb = 0; jb < 32; ++jb) {
    if (jb + 1 < 32)
      __builtin_prefetch(Kp + (size_t)((jb + 1) * 32 + nl) * kD, 0, 0);
    v8f acc0 = {}, acc1 = {};
#pragma unroll
    for (int dc = 0; dc < 2; ++dc) {
      const unsigned short* b0 = Kp + (size_t)(jb * 32 + nl) * kD + dc * 32 + kb;
      const unsigned short* b1 = Kp + (size_t)(jb * 32 + 16 + nl) * kD + dc * 32 + kb;
      acc0 = wmma_bf16(qa[dc], bfrag(b0), acc0);
      acc1 = wmma_bf16(qa[dc], bfrag(b1), acc1);
    }
#pragma unroll
    for (int r = 0; r < 8; ++r) {
      float v = fmaxf(acc0[r], acc1[r]);
      v = fmaxf(v, __shfl_xor(v, 1, 32));
      v = fmaxf(v, __shfl_xor(v, 2, 32));
      v = fmaxf(v, __shfl_xor(v, 4, 32));
      v = fmaxf(v, __shfl_xor(v, 8, 32));
      const float nm = fmaxf(rm[r], v);
      float s = __expf(acc0[r] - nm) + __expf(acc1[r] - nm);
      s += __shfl_xor(s, 1, 32);
      s += __shfl_xor(s, 2, 32);
      s += __shfl_xor(s, 4, 32);
      s += __shfl_xor(s, 8, 32);
      rl[r] = rl[r] * __expf(rm[r] - nm) + s;
      rm[r] = nm;
    }
  }

  if (nl == 0) {
    const int half8 = (lane < 16) ? 0 : 8;
#pragma unroll
    for (int r = 0; r < 8; ++r) {
      const int row = bh * kN + row0 + r + half8;
      m_ws[row] = rm[r];
      l_ws[row] = rl[r];
    }
  }
}

// ---------------------------------------------------------------------------
// Kernel 4: softmax pass 2 + attend. Recompute S, P = exp(S-m)/l, write A,
// then O1 += P@V1, O2 += P@V2 via WMMA (P re-tiled through LDS, V staged
// transposed in LDS, row stride 40 keeps every fragment read 16B aligned).
// ---------------------------------------------------------------------------
__global__ __launch_bounds__(256) void k_pass2(
    const unsigned short* __restrict__ proj,
    const float* __restrict__ m_ws, const float* __restrict__ l_ws,
    float* __restrict__ Aout,
    unsigned short* __restrict__ E1, unsigned short* __restrict__ E2) {
  const int bh = blockIdx.x >> 3;
  const int rb = blockIdx.x & 7;
  const int b  = bh >> 4;
  const int h  = bh & 15;
  const unsigned short* Qp  = proj + (size_t)(128 + bh) * (kN * kD);
  const unsigned short* Kp  = proj + (size_t)(bh) * (kN * kD);
  const unsigned short* V1p = proj + (size_t)(256 + bh) * (kN * kD);
  const unsigned short* V2p = proj + (size_t)(384 + bh) * (kN * kD);

  __shared__ __align__(16) unsigned short v1t[64][40];  // V1 tile^T [d][j]
  __shared__ __align__(16) unsigned short v2t[64][40];
  __shared__ __align__(16) unsigned short plds[8][16][32];

  const int lane  = threadIdx.x & 31;
  const int wave  = threadIdx.x >> 5;
  const int nl    = lane & 15;
  const int half8 = (lane < 16) ? 0 : 8;
  const int ab    = (lane < 16) ? 0 : 16;
  const int kb    = (lane < 16) ? 0 : 16;
  const int row0  = rb * 128 + wave * 16;

  v16bf qa[2];
#pragma unroll
  for (int dc = 0; dc < 2; ++dc)
    qa[dc] = afrag(Qp + (size_t)(row0 + nl) * kD + dc * 32 + ab);

  float rm[8], rli[8];
#pragma unroll
  for (int r = 0; r < 8; ++r) {
    const int row = bh * kN + row0 + r + half8;
    rm[r]  = m_ws[row];
    rli[r] = 1.f / l_ws[row];
  }

  v8f o1[4] = {}, o2[4] = {};

  for (int jb = 0; jb < 32; ++jb) {
    __syncthreads();  // previous iteration done with LDS
    for (int i = threadIdx.x; i < 2048; i += 256) {
      const int j = i >> 6;
      const int d = i & 63;
      v1t[d][j] = V1p[(size_t)(jb * 32 + j) * kD + d];
      v2t[d][j] = V2p[(size_t)(jb * 32 + j) * kD + d];
    }
    if (jb + 1 < 32)
      __builtin_prefetch(Kp + (size_t)((jb + 1) * 32 + nl) * kD, 0, 0);
    __syncthreads();

    v8f acc0 = {}, acc1 = {};
#pragma unroll
    for (int dc = 0; dc < 2; ++dc) {
      const unsigned short* b0 = Kp + (size_t)(jb * 32 + nl) * kD + dc * 32 + kb;
      const unsigned short* b1 = Kp + (size_t)(jb * 32 + 16 + nl) * kD + dc * 32 + kb;
      acc0 = wmma_bf16(qa[dc], bfrag(b0), acc0);
      acc1 = wmma_bf16(qa[dc], bfrag(b1), acc1);
    }

#pragma unroll
    for (int r = 0; r < 8; ++r) {
      const int rr = r + half8;
      const float p0 = __expf(acc0[r] - rm[r]) * rli[r];
      const float p1 = __expf(acc1[r] - rm[r]) * rli[r];
      const size_t arow = (size_t)(bh * kN + row0 + rr) * kN + jb * 32;
      Aout[arow + nl]      = p0;
      Aout[arow + 16 + nl] = p1;
      plds[wave][rr][nl]      = f2bfu(p0);
      plds[wave][rr][16 + nl] = f2bfu(p1);
    }
    __syncthreads();

    const v16bf pa = afrag(&plds[wave][nl][ab]);
    const int jo = (lane < 16) ? 0 : 16;
#pragma unroll
    for (int c = 0; c < 4; ++c) {
      o1[c] = wmma_bf16(pa, bfrag(&v1t[c * 16 + nl][jo]), o1[c]);
      o2[c] = wmma_bf16(pa, bfrag(&v2t[c * 16 + nl][jo]), o2[c]);
    }
  }

#pragma unroll
  for (int c = 0; c < 4; ++c) {
#pragma unroll
    for (int r = 0; r < 8; ++r) {
      const int n = row0 + r + half8;
      const size_t ei = ((size_t)(b * kN + n)) * (kH * kD) + h * kD + c * 16 + nl;
      E1[ei] = f2bfu(o1[c][r]);
      E2[ei] = f2bfu(o2[c][r]);
    }
  }
}

// ---------------------------------------------------------------------------
// Kernel 5: out = relu(E @ W1^T + b1)   M=8192 N'=512 K=1024, twice.
// ---------------------------------------------------------------------------
__global__ __launch_bounds__(256) void k_fc(
    const unsigned short* __restrict__ E1, const unsigned short* __restrict__ E2,
    const unsigned short* __restrict__ W1b, const float* __restrict__ b1,
    float* __restrict__ out1, float* __restrict__ out2) {
  const int which = blockIdx.x >> 9;
  const int rem   = blockIdx.x & 511;
  const int mt    = rem >> 3;
  const int ct    = rem & 7;
  const unsigned short* E = which ? E2 : E1;
  float* out = which ? out2 : out1;

  const int lane  = threadIdx.x & 31;
  const int wave  = threadIdx.x >> 5;
  const int nl    = lane & 15;
  const int half8 = (lane < 16) ? 0 : 8;
  const int ab    = (lane < 16) ? 0 : 16;
  const int kb    = (lane < 16) ? 0 : 16;
  const int row0  = mt * 128 + wave * 16;
  const int n0    = ct * 64;

  v8f acc[4] = {};
  for (int kc = 0; kc < 32; ++kc) {
    const v16bf a = afrag(E + (size_t)(row0 + nl) * (kH * kD) + kc * 32 + ab);
    const int koff = kc * 32 + kb;
#pragma unroll
    for (int c = 0; c < 4; ++c) {
      acc[c] = wmma_bf16(
          a, bfrag(W1b + (size_t)(n0 + c * 16 + nl) * (kH * kD) + koff), acc[c]);
    }
  }
#pragma unroll
  for (int c = 0; c < 4; ++c) {
    const int n = n0 + c * 16 + nl;
    const float bc = b1[n];
#pragma unroll
    for (int r = 0; r < 8; ++r) {
      const float v = fmaxf(acc[c][r] + bc, 0.f);
      out[(size_t)(row0 + r + half8) * kIN + n] = v;
    }
  }
}

// ---------------------------------------------------------------------------
extern "C" void kernel_launch(void* const* d_in, const int* in_sizes, int n_in,
                              void* d_out, int out_size, void* d_ws, size_t ws_size,
                              hipStream_t stream) {
  const float* m1   = (const float*)d_in[0];
  const float* m2   = (const float*)d_in[1];
  const float* Wq   = (const float*)d_in[2];
  const float* bq   = (const float*)d_in[3];
  const float* Wk   = (const float*)d_in[4];
  const float* bk   = (const float*)d_in[5];
  const float* Wv   = (const float*)d_in[6];
  const float* bv   = (const float*)d_in[7];
  const float* qn_w = (const float*)d_in[8];
  const float* qn_b = (const float*)d_in[9];
  const float* kn_w = (const float*)d_in[10];
  const float* kn_b = (const float*)d_in[11];
  const float* vn_w = (const float*)d_in[12];
  const float* vn_b = (const float*)d_in[13];
  const float* W1   = (const float*)d_in[14];
  const float* b1   = (const float*)d_in[15];
  // d_in[16] = comb (unused by the reference math)

  char* ws = (char*)d_ws;
  unsigned short* proj  = (unsigned short*)(ws + 0);
  float*          stats = (float*)(ws + 67108864);
  float*          m_ws  = (float*)(ws + 67112960);
  float*          l_ws  = (float*)(ws + 67637248);
  unsigned short* E1    = (unsigned short*)(ws + 68161536);
  unsigned short* E2    = (unsigned short*)(ws + 84938752);
  unsigned short* m1b   = (unsigned short*)(ws + 101715968);
  unsigned short* m2b   = (unsigned short*)(ws + 110104576);
  unsigned short* Wkb   = (unsigned short*)(ws + 118493184);
  unsigned short* Wqb   = (unsigned short*)(ws + 119541760);
  unsigned short* Wvb   = (unsigned short*)(ws + 120590336);
  unsigned short* W1b   = (unsigned short*)(ws + 121638912);

  float* out  = (float*)d_out;
  float* out1 = out;
  float* out2 = out + (size_t)kB * kN * kIN;       // 4194304
  float* Aout = out + (size_t)2 * kB * kN * kIN;   // 8388608

  // fp32 -> bf16 operand pre-conversion (each thread converts 4 elements)
  k_cvt<<<4096, 256, 0, stream>>>(m1, m1b);   // 4,194,304 elems
  k_cvt<<<4096, 256, 0, stream>>>(m2, m2b);
  k_cvt<<<512, 256, 0, stream>>>(Wk, Wkb);    // 524,288 elems
  k_cvt<<<512, 256, 0, stream>>>(Wq, Wqb);
  k_cvt<<<512, 256, 0, stream>>>(Wv, Wvb);
  k_cvt<<<512, 256, 0, stream>>>(W1, W1b);

  k_proj_ln<<<512, 256, 0, stream>>>(m1b, m2b, Wqb, Wkb, Wvb, bq, bk, bv,
                                     proj, stats);
  k_ln_apply<<<32768, 256, 0, stream>>>(proj, stats, qn_w, qn_b, kn_w, kn_b,
                                        vn_w, vn_b);
  k_pass1<<<1024, 256, 0, stream>>>(proj, m_ws, l_ws);
  k_pass2<<<1024, 256, 0, stream>>>(proj, m_ws, l_ws, Aout, E1, E2);
  k_fc<<<1024, 256, 0, stream>>>(E1, E2, W1b, b1, out1, out2);
}
